// CustomEmbeddings_75342316307026
// MI455X (gfx1250) — matrix operand order repaired and checked by reading, not compiled, not run on hardware
//
#include <hip/hip_runtime.h>
#include <hip/hip_bf16.h>
#include <math.h>

typedef __attribute__((ext_vector_type(16))) __bf16 v16bf;
typedef __attribute__((ext_vector_type(8)))  __bf16 v8bf;
typedef __attribute__((ext_vector_type(8)))  float  v8f;
typedef __attribute__((ext_vector_type(4)))  unsigned int v4u;
typedef __attribute__((ext_vector_type(8)))  int v8i;
typedef __attribute__((ext_vector_type(4)))  int v4i;

#define OLD_V   50257
#define NEW_V   53257
#define NEWROWS (NEW_V - OLD_V)   // 3000
#define DIM     2048
#define T_TOK   32768             // B*S
#define NNUM    4096
#define HID     8192              // 4*D
#define LSTR    40                // bf16 per LDS tile row (64B data + 16B pad)

// ---------------------------------------------------------------------------
// Kernel 1: embedding gather + range-select, float4 vectorized (pure BW bound)
// ---------------------------------------------------------------------------
__global__ __launch_bounds__(256) void emb_gather_kernel(
    const int*   __restrict__ ids,
    const float* __restrict__ orig_emb,
    const float* __restrict__ new_emb,
    float*       __restrict__ out) {
  int idx = blockIdx.x * 256 + threadIdx.x;   // one float4 per thread
  int t = idx >> 9;                           // D/4 = 512 float4 per row
  int c = idx & 511;
  int id = __ldg(&ids[t]);
  const float4* src;
  if (id >= OLD_V) {
    int sel = id - OLD_V;
    if (sel > NEWROWS - 1) sel = NEWROWS - 1;
    src = (const float4*)(new_emb + (size_t)sel * DIM);
  } else {
    src = (const float4*)(orig_emb + (size_t)id * DIM);
  }
  ((float4*)out)[idx] = src[c];
}

// ---------------------------------------------------------------------------
// Kernel 2a: h = gelu(feats @ W1 + b1) as bf16 [NNUM, HID], computed ONCE.
// One thread computes 8 consecutive k of one row; 16B packed bf16 store.
// ---------------------------------------------------------------------------
__global__ __launch_bounds__(256) void h_compute_kernel(
    const float* __restrict__ num_values,
    const int*   __restrict__ num_units,
    const float* __restrict__ unit_emb,
    const float* __restrict__ W1,
    const float* __restrict__ b1,
    __bf16*      __restrict__ h) {
  int idx = blockIdx.x * 256 + threadIdx.x;   // NNUM * (HID/8) chunks
  int m  = idx >> 10;                         // HID/8 = 1024 chunks per row
  int kc = (idx & 1023) << 3;
  float f0 = num_values[m];
  int   u  = num_units[m];
  float f1 = unit_emb[u * 2 + 0];
  float f2 = unit_emb[u * 2 + 1];

  float4 w0a = *(const float4*)(W1 + kc);
  float4 w0b = *(const float4*)(W1 + kc + 4);
  float4 w1a = *(const float4*)(W1 + HID + kc);
  float4 w1b = *(const float4*)(W1 + HID + kc + 4);
  float4 w2a = *(const float4*)(W1 + 2 * HID + kc);
  float4 w2b = *(const float4*)(W1 + 2 * HID + kc + 4);
  float4 bba = *(const float4*)(b1 + kc);
  float4 bbb = *(const float4*)(b1 + kc + 4);

  float x[8];
  x[0] = fmaf(f0, w0a.x, fmaf(f1, w1a.x, fmaf(f2, w2a.x, bba.x)));
  x[1] = fmaf(f0, w0a.y, fmaf(f1, w1a.y, fmaf(f2, w2a.y, bba.y)));
  x[2] = fmaf(f0, w0a.z, fmaf(f1, w1a.z, fmaf(f2, w2a.z, bba.z)));
  x[3] = fmaf(f0, w0a.w, fmaf(f1, w1a.w, fmaf(f2, w2a.w, bba.w)));
  x[4] = fmaf(f0, w0b.x, fmaf(f1, w1b.x, fmaf(f2, w2b.x, bbb.x)));
  x[5] = fmaf(f0, w0b.y, fmaf(f1, w1b.y, fmaf(f2, w2b.y, bbb.y)));
  x[6] = fmaf(f0, w0b.z, fmaf(f1, w1b.z, fmaf(f2, w2b.z, bbb.z)));
  x[7] = fmaf(f0, w0b.w, fmaf(f1, w1b.w, fmaf(f2, w2b.w, bbb.w)));

  v8bf r;
#pragma unroll
  for (int i = 0; i < 8; ++i) {
    float gl = 0.5f * x[i] * (1.0f + erff(x[i] * 0.7071067811865475f));
    r[i] = (__bf16)gl;
  }
  *(v8bf*)(h + (size_t)m * HID + kc) = r;
}

// ---------------------------------------------------------------------------
// Kernel 2b: GEMM h[4096,8192](bf16) @ W2[8192,2048] -> scatter-add + b2.
// Block tile 128x128, K-step 32.  A tile DMA'd by the Tensor Data Mover
// (tensor_load_to_lds, 2-D D#, LDS pad 16B per 64B row -> 80B row stride),
// B tile converted fp32->bf16 by the threads (overlaps the TDM transfer).
// ---------------------------------------------------------------------------
__global__ __launch_bounds__(256) void mlp_wmma_tdm_kernel(
    const int*    __restrict__ num_positions,
    const __bf16* __restrict__ h,
    const float*  __restrict__ W2,
    const float*  __restrict__ b2,
    float*        __restrict__ out) {
  __shared__ __bf16 As[128 * LSTR];
  __shared__ __bf16 Bs[128 * LSTR];
  __shared__ int    poss[128];

  const int tid     = threadIdx.x;
  const int m_block = blockIdx.y * 128;
  const int n_block = blockIdx.x * 128;

  if (tid < 128) poss[tid] = num_positions[m_block + tid];

  const int wave = tid >> 5;
  const int lane = tid & 31;
  const int wy   = wave >> 1;     // 0..3 : 32-row slab
  const int wx   = wave & 1;      // 0..1 : 64-col slab
  const int lm   = lane & 15;
  const int g    = lane >> 4;

  v8f acc[2][4];
#pragma unroll
  for (int i = 0; i < 2; ++i)
#pragma unroll
    for (int j = 0; j < 4; ++j)
#pragma unroll
      for (int e = 0; e < 8; ++e) acc[i][j][e] = 0.0f;

  // ---- TDM descriptor group 1 (constant): data_size=2B, pad_enable,
  // pad_interval=3 (16 DW = 64B), pad_amount=3 (4 DW = 16B),
  // tensor_dim0=8192, tensor_dim1=4096, tile_dim0=32, tile_dim1=128,
  // tensor_dim0_stride=8192.
  const v8i g1 = { (int)((1u << 16) | (1u << 20) | (3u << 22) | (3u << 25)),
                   (int)(0x2000u << 16),   // tensor_dim0 = 8192 -> bits[79:48]
                   (int)(0x1000u << 16),   // tensor_dim1 = 4096 -> bits[111:80]
                   (int)(32u << 16),       // tile_dim0 = 32     -> bits[127:112]
                   128,                    // tile_dim1 = 128    -> bits[143:128]
                   8192,                   // tensor_dim0_stride -> bits[207:160]
                   0, 0 };
  const v4i zero4 = { 0, 0, 0, 0 };
  const v8i zero8 = { 0, 0, 0, 0, 0, 0, 0, 0 };
  const unsigned lds_base =
      (unsigned)(unsigned long long)(const void*)&As[0];
  const unsigned long long hbase =
      (unsigned long long)h + (unsigned long long)m_block * HID * 2ull;

  const int b_nn = tid & 127;     // N column for B fill
  const int b_kh = tid >> 7;      // K half (0/1) for B fill

  for (int k0 = 0; k0 < HID; k0 += 32) {
    __syncthreads();
    // ---- issue TDM for the A tile (wave 0; EXEC-independent, per-wave op)
    if (wave == 0) {
      unsigned long long ga = hbase + (unsigned long long)k0 * 2ull;
      v4u g0 = { 1u,                                   // count=1, user mode
                 lds_base,                             // lds_addr
                 (unsigned)ga,                         // global_addr[31:0]
                 (unsigned)((ga >> 32) & 0x1FFFFFFull) | (2u << 30) }; // type=2
      __builtin_amdgcn_tensor_load_to_lds(g0, g1, zero4, zero4, zero8, 0);
    }
    // ---- B tile fill overlaps the TDM DMA: W2[k0..+32, n0..+128] as [n][k]
#pragma unroll
    for (int j = 0; j < 16; ++j) {
      int kk = b_kh * 16 + j;
      Bs[b_nn * LSTR + kk] = (__bf16)W2[(size_t)(k0 + kk) * DIM + n_block + b_nn];
    }
    if (wave == 0) __builtin_amdgcn_s_wait_tensorcnt(0);
    __syncthreads();

    // ---- fragments + WMMA
    v16bf afr[2];
#pragma unroll
    for (int mi = 0; mi < 2; ++mi) {
      int m = wy * 32 + mi * 16 + lm;
      v8bf lo = *(const v8bf*)&As[m * LSTR + g * 8];
      v8bf hi = *(const v8bf*)&As[m * LSTR + 16 + g * 8];
      v16bf a;
#pragma unroll
      for (int e = 0; e < 8; ++e) { a[e] = lo[e]; a[e + 8] = hi[e]; }
      afr[mi] = a;
    }
#pragma unroll
    for (int ni = 0; ni < 4; ++ni) {
      int n = wx * 64 + ni * 16 + lm;
      v8bf lo = *(const v8bf*)&Bs[n * LSTR + g * 8];
      v8bf hi = *(const v8bf*)&Bs[n * LSTR + 16 + g * 8];
      v16bf b;
#pragma unroll
      for (int e = 0; e < 8; ++e) { b[e] = lo[e]; b[e + 8] = hi[e]; }
#pragma unroll
      for (int mi = 0; mi < 2; ++mi) {
        acc[mi][ni] = __builtin_amdgcn_wmma_f32_16x16x32_bf16(
            false, afr[mi], false, b, (short)0, acc[mi][ni], false, false);
      }
    }
  }

  // ---- epilogue: scatter-add with bias.  D layout: (M = g*8 + i, N = lm)
#pragma unroll
  for (int ni = 0; ni < 4; ++ni) {
    int n = n_block + wx * 64 + ni * 16 + lm;
    float bias = b2[n];
#pragma unroll
    for (int mi = 0; mi < 2; ++mi) {
#pragma unroll
      for (int i = 0; i < 8; ++i) {
        int mloc = wy * 32 + mi * 16 + g * 8 + i;
        int pos  = poss[mloc];
        float* p = out + (size_t)pos * DIM + n;
        *p += acc[mi][ni][i] + bias;
      }
    }
  }
}

// ---------------------------------------------------------------------------
// Fallback (ws too small): fused gelu-in-the-GEMM version (Round-0 kernel).
// ---------------------------------------------------------------------------
__global__ __launch_bounds__(256) void mlp_wmma_fused_kernel(
    const int*   __restrict__ num_positions,
    const float* __restrict__ num_values,
    const int*   __restrict__ num_units,
    const float* __restrict__ unit_emb,
    const float* __restrict__ W1,
    const float* __restrict__ b1,
    const float* __restrict__ W2,
    const float* __restrict__ b2,
    float*       __restrict__ out) {
  __shared__ __bf16 As[128 * LSTR];
  __shared__ __bf16 Bs[128 * LSTR];
  __shared__ float  f0s[128], f1s[128], f2s[128];
  __shared__ int    poss[128];

  const int tid     = threadIdx.x;
  const int m_block = blockIdx.y * 128;
  const int n_block = blockIdx.x * 128;

  if (tid < 128) {
    int mg = m_block + tid;
    f0s[tid]  = num_values[mg];
    int u     = num_units[mg];
    f1s[tid]  = unit_emb[u * 2 + 0];
    f2s[tid]  = unit_emb[u * 2 + 1];
    poss[tid] = num_positions[mg];
  }

  const int wave = tid >> 5;
  const int lane = tid & 31;
  const int wy   = wave >> 1;
  const int wx   = wave & 1;
  const int lm   = lane & 15;
  const int g    = lane >> 4;

  v8f acc[2][4];
#pragma unroll
  for (int i = 0; i < 2; ++i)
#pragma unroll
    for (int j = 0; j < 4; ++j)
#pragma unroll
      for (int e = 0; e < 8; ++e) acc[i][j][e] = 0.0f;

  const int a_kk = tid & 31;
  const int a_mg = tid >> 5;
  const int b_nn = tid & 127;
  const int b_kh = tid >> 7;

  for (int k0 = 0; k0 < HID; k0 += 32) {
    __syncthreads();
    {
      int k = k0 + a_kk;
      float w0 = W1[k];
      float w1 = W1[HID + k];
      float w2 = W1[2 * HID + k];
      float bb = b1[k];
#pragma unroll
      for (int r = 0; r < 16; ++r) {
        int m = a_mg * 16 + r;
        float x  = fmaf(f0s[m], w0, fmaf(f1s[m], w1, fmaf(f2s[m], w2, bb)));
        float gl = 0.5f * x * (1.0f + erff(x * 0.7071067811865475f));
        As[m * LSTR + a_kk] = (__bf16)gl;
      }
    }
#pragma unroll
    for (int j = 0; j < 16; ++j) {
      int kk = b_kh * 16 + j;
      Bs[b_nn * LSTR + kk] = (__bf16)W2[(size_t)(k0 + kk) * DIM + n_block + b_nn];
    }
    __syncthreads();

    v16bf afr[2];
#pragma unroll
    for (int mi = 0; mi < 2; ++mi) {
      int m = wy * 32 + mi * 16 + lm;
      v8bf lo = *(const v8bf*)&As[m * LSTR + g * 8];
      v8bf hi = *(const v8bf*)&As[m * LSTR + 16 + g * 8];
      v16bf a;
#pragma unroll
      for (int e = 0; e < 8; ++e) { a[e] = lo[e]; a[e + 8] = hi[e]; }
      afr[mi] = a;
    }
#pragma unroll
    for (int ni = 0; ni < 4; ++ni) {
      int n = wx * 64 + ni * 16 + lm;
      v8bf lo = *(const v8bf*)&Bs[n * LSTR + g * 8];
      v8bf hi = *(const v8bf*)&Bs[n * LSTR + 16 + g * 8];
      v16bf b;
#pragma unroll
      for (int e = 0; e < 8; ++e) { b[e] = lo[e]; b[e + 8] = hi[e]; }
#pragma unroll
      for (int mi = 0; mi < 2; ++mi) {
        acc[mi][ni] = __builtin_amdgcn_wmma_f32_16x16x32_bf16(
            false, afr[mi], false, b, (short)0, acc[mi][ni], false, false);
      }
    }
  }

#pragma unroll
  for (int ni = 0; ni < 4; ++ni) {
    int n = n_block + wx * 64 + ni * 16 + lm;
    float bias = b2[n];
#pragma unroll
    for (int mi = 0; mi < 2; ++mi) {
#pragma unroll
      for (int i = 0; i < 8; ++i) {
        int mloc = wy * 32 + mi * 16 + g * 8 + i;
        int pos  = poss[mloc];
        float* p = out + (size_t)pos * DIM + n;
        *p += acc[mi][ni][i] + bias;
      }
    }
  }
}

// ---------------------------------------------------------------------------
extern "C" void kernel_launch(void* const* d_in, const int* in_sizes, int n_in,
                              void* d_out, int out_size, void* d_ws, size_t ws_size,
                              hipStream_t stream) {
  (void)in_sizes; (void)n_in; (void)out_size;

  const int*   input_ids     = (const int*)  d_in[0];
  const int*   num_positions = (const int*)  d_in[1];
  const float* num_values    = (const float*)d_in[2];
  const int*   num_units     = (const int*)  d_in[3];
  const float* orig_emb      = (const float*)d_in[4];
  const float* new_emb       = (const float*)d_in[5];
  const float* unit_emb      = (const float*)d_in[6];
  const float* W1            = (const float*)d_in[7];
  const float* b1            = (const float*)d_in[8];
  const float* W2            = (const float*)d_in[9];
  const float* b2            = (const float*)d_in[10];
  float*       out           = (float*)d_out;

  // Kernel 1: 16,777,216 float4 -> 65536 blocks x 256
  emb_gather_kernel<<<(T_TOK * (DIM / 4)) / 256, 256, 0, stream>>>(
      input_ids, orig_emb, new_emb, out);

  dim3 grid(DIM / 128, NNUM / 128);
  const size_t h_bytes = (size_t)NNUM * HID * sizeof(__bf16);   // 64 MB

  if (ws_size >= h_bytes) {
    __bf16* h = (__bf16*)d_ws;
    // Kernel 2a: NNUM * HID / 8 chunks -> 16384 blocks x 256
    h_compute_kernel<<<(NNUM * (HID / 8)) / 256, 256, 0, stream>>>(
        num_values, num_units, unit_emb, W1, b1, h);
    // Kernel 2b: TDM + WMMA GEMM with scatter-add epilogue
    mlp_wmma_tdm_kernel<<<grid, 256, 0, stream>>>(
        num_positions, h, W2, b2, out);
  } else {
    mlp_wmma_fused_kernel<<<grid, 256, 0, stream>>>(
        num_positions, num_values, num_units, unit_emb, W1, b1, W2, b2, out);
  }
}
